// BitLayer_6090263626027
// MI455X (gfx1250) — 8-bit matrix operand order repaired and checked by reading, compiled, not run on hardware
//
#include <hip/hip_runtime.h>
#include <hip/hip_bf16.h>
#include <stdint.h>

typedef __attribute__((ext_vector_type(8))) int v8i;

#define BATCH   32
#define IN_DIM  512
#define NUM_OUT 128
#define BITS    64

__global__ __launch_bounds__(256) void pack_x_kernel(const int* __restrict__ x,
                                                     uint8_t* __restrict__ ap) {
    uint32_t tid = blockIdx.x * 256u + threadIdx.x;
    uint32_t dg  = tid & 127u;
    uint32_t b   = (tid >> 7) & 31u;
    uint32_t s   = tid >> 12;
    uint32_t d0  = dg << 2;
    const int* p = x + ((size_t)(b * IN_DIM + d0) * BITS + s);
    uint32_t v0 = (uint32_t)p[0 * BITS] & 1u;
    uint32_t v1 = (uint32_t)p[1 * BITS] & 1u;
    uint32_t v2 = (uint32_t)p[2 * BITS] & 1u;
    uint32_t v3 = (uint32_t)p[3 * BITS] & 1u;
    ((uint32_t*)ap)[tid] = v0 | (v1 << 8) | (v2 << 16) | (v3 << 24);
}

__global__ __launch_bounds__(256) void pack_k_kernel(const int* __restrict__ k,
                                                     uint8_t* __restrict__ bp) {
    uint32_t tid = blockIdx.x * 256u + threadIdx.x;
    uint32_t dg  = tid & 127u;
    uint32_t o   = (tid >> 7) & 127u;
    uint32_t s   = tid >> 14;
    uint32_t d0  = dg << 2;
    const int* p = k + ((size_t)(d0 * NUM_OUT + o) * BITS + s);
    const uint32_t dstr = NUM_OUT * BITS;
    uint32_t v0 = (uint32_t)p[0 * dstr] & 1u;
    uint32_t v1 = (uint32_t)p[1 * dstr] & 1u;
    uint32_t v2 = (uint32_t)p[2 * dstr] & 1u;
    uint32_t v3 = (uint32_t)p[3 * dstr] & 1u;
    ((uint32_t*)bp)[tid] = v0 | (v1 << 8) | (v2 << 16) | (v3 << 24);
}

__global__ __launch_bounds__(256) void bitmm_wmma_kernel(const uint8_t* __restrict__ ap,
                                                         const uint8_t* __restrict__ bp,
                                                         float* __restrict__ out) {
    const uint32_t lane   = threadIdx.x & 31u;
    const uint32_t ntile  = threadIdx.x >> 5;
    const uint32_t s      = blockIdx.x >> 1;
    const uint32_t mtile  = blockIdx.x & 1u;
    const uint32_t lane16 = lane & 15u;
    const uint32_t hi     = lane >> 4;

    const uint8_t* arow = ap + (size_t)((s * BATCH)   + mtile * 16 + lane16) * IN_DIM;
    const uint8_t* brow = bp + (size_t)((s * NUM_OUT) + ntile * 16 + lane16) * IN_DIM;

    const uint32_t ahi8  = hi * 16;
    const uint32_t bhi16 = hi * 16;

    v8i c = {};
#pragma unroll
    for (int kc = 0; kc < 8; ++kc) {
        const uint32_t k0 = (uint32_t)kc * 64u;
        v8i a, b;
        uint2 a0 = *(const uint2*)(arow + k0 + ahi8 + 0);
        uint2 a1 = *(const uint2*)(arow + k0 + ahi8 + 8);
        uint2 a2 = *(const uint2*)(arow + k0 + ahi8 + 32);
        uint2 a3 = *(const uint2*)(arow + k0 + ahi8 + 40);
        a[0] = (int)a0.x; a[1] = (int)a0.y;
        a[2] = (int)a1.x; a[3] = (int)a1.y;
        a[4] = (int)a2.x; a[5] = (int)a2.y;
        a[6] = (int)a3.x; a[7] = (int)a3.y;
        uint4 b0 = *(const uint4*)(brow + k0 + bhi16 + 0);
        uint4 b1 = *(const uint4*)(brow + k0 + bhi16 + 32);
        b[0] = (int)b0.x; b[1] = (int)b0.y; b[2] = (int)b0.z; b[3] = (int)b0.w;
        b[4] = (int)b1.x; b[5] = (int)b1.y; b[6] = (int)b1.z; b[7] = (int)b1.w;

        c = __builtin_amdgcn_wmma_i32_16x16x64_iu8(false, a, false, b, c, false, false);
    }

    const uint32_t o = ntile * 16 + lane16;
#pragma unroll
    for (int r = 0; r < 8; ++r) {
        const uint32_t bb = mtile * 16 + hi * 8 + (uint32_t)r;
        out[((size_t)bb * NUM_OUT + o) * BITS + s] = (c[r] > 0) ? 1.0f : 0.0f;
    }
}

extern "C" void kernel_launch(void* const* d_in, const int* in_sizes, int n_in,
                              void* d_out, int out_size, void* d_ws, size_t ws_size,
                              hipStream_t stream) {
    const int* x = (const int*)d_in[0];
    const int* k = (const int*)d_in[1];
    float* out   = (float*)d_out;

    uint8_t* ap = (uint8_t*)d_ws;
    uint8_t* bp = ap + (size_t)BATCH * IN_DIM * BITS;

    pack_x_kernel<<<1024, 256, 0, stream>>>(x, ap);
    pack_k_kernel<<<4096, 256, 0, stream>>>(k, bp);
    bitmm_wmma_kernel<<<128, 256, 0, stream>>>(ap, bp, out);
}
